// HeteroSAGENet_4604204941984
// MI455X (gfx1250) — matrix operand order repaired and checked by reading, not compiled, not run on hardware
//
#include <hip/hip_runtime.h>
#include <hip/hip_bf16.h>

typedef __attribute__((ext_vector_type(16))) _Float16 v16h;
typedef __attribute__((ext_vector_type(8)))  float    v8f;

#define N_FM 100000
#define N_TP 4096
#define N_SM 100000
#define HID  128

// ---------------- utility kernels ----------------

__global__ void k_zero(float* p, unsigned long long n) {
  unsigned long long i = (unsigned long long)blockIdx.x * blockDim.x + threadIdx.x;
  unsigned long long stride = (unsigned long long)gridDim.x * blockDim.x;
  for (; i < n; i += stride) p[i] = 0.0f;
}

__global__ void k_pe(const float* __restrict__ pe_table,
                     const float* __restrict__ vol,
                     float* __restrict__ pe) {
  int t = blockIdx.x * blockDim.x + threadIdx.x;
  if (t < N_TP * 32) pe[t] = pe_table[t] * vol[t >> 5];
}

// Edge scatter: (1<<lshift) lanes per edge, each lane moves one float4.
// acc is [n_dst, F] with F = 4<<lshift; cnt counts edges per dst.
__global__ void k_edge(const float* __restrict__ feat,
                       const int* __restrict__ src,
                       const int* __restrict__ dst,
                       int E, int lshift,
                       float* __restrict__ acc,
                       float* __restrict__ cnt) {
  unsigned t = blockIdx.x * blockDim.x + threadIdx.x;
  unsigned e = t >> lshift;
  if (e >= (unsigned)E) return;
  unsigned sub = t & ((1u << lshift) - 1u);
  int F = 4 << lshift;
  int s = src[e], d = dst[e];
  float4 v = ((const float4*)(feat + (size_t)s * F))[sub];
  float* o = acc + (size_t)d * F + sub * 4;
  atomicAdd(o + 0, v.x);
  atomicAdd(o + 1, v.y);
  atomicAdd(o + 2, v.z);
  atomicAdd(o + 3, v.w);
  if (sub == 0) atomicAdd(cnt + d, 1.0f);
}

// acc[t] /= max(cnt[row],1); row = t >> fshift
__global__ void k_div(float* __restrict__ acc, const float* __restrict__ cnt,
                      unsigned n_elems, int fshift) {
  unsigned t = blockIdx.x * blockDim.x + threadIdx.x;
  if (t < n_elems) {
    float c = cnt[t >> fshift];
    acc[t] *= (c > 1.0f) ? (1.0f / c) : 1.0f;
  }
}

// ---------------- weight packing (transposed fp16, fused K) ----------------

// fm: K=192 = [qoq_Wl(64) | cp_Wl(32) | rb_Wl(32) | qoq_Wr+cp_Wr+rb_Wr(64)]
__global__ void k_pack_fm(const float* __restrict__ qWl, const float* __restrict__ cWl,
                          const float* __restrict__ rWl,
                          const float* __restrict__ qWr, const float* __restrict__ cWr,
                          const float* __restrict__ rWr,
                          const float* __restrict__ qbl, const float* __restrict__ cbl,
                          const float* __restrict__ rbl,
                          _Float16* __restrict__ BT, float* __restrict__ bias) {
  int t = blockIdx.x * blockDim.x + threadIdx.x;
  if (t < 128 * 192) {
    int c = t / 192, k = t % 192;
    float v;
    if (k < 64)       v = qWl[k * HID + c];
    else if (k < 96)  v = cWl[(k - 64) * HID + c];
    else if (k < 128) v = rWl[(k - 96) * HID + c];
    else              v = qWr[(k - 128) * HID + c] + cWr[(k - 128) * HID + c] +
                          rWr[(k - 128) * HID + c];
    BT[c * 192 + k] = (_Float16)v;
  }
  if (t < 128) bias[t] = qbl[t] + cbl[t] + rbl[t];
}

// generic 2-region pack: K = Fl+Fr, BT is [128][K]
__global__ void k_pack2(const float* __restrict__ Wl, int Fl,
                        const float* __restrict__ Wr, int Fr,
                        _Float16* __restrict__ BT) {
  int K = Fl + Fr;
  int t = blockIdx.x * blockDim.x + threadIdx.x;
  if (t < 128 * K) {
    int c = t / K, k = t % K;
    float v = (k < Fl) ? Wl[k * HID + c] : Wr[(k - Fl) * HID + c];
    BT[c * K + k] = (_Float16)v;
  }
}

// ---------------- fused GEMM + LayerNorm + ReLU + column-mean ----------------

__device__ __forceinline__ void cvt8(_Float16* dst, const float* p) {
  const float4* q = (const float4*)p;
  float4 x = q[0], y = q[1];
  dst[0] = (_Float16)x.x; dst[1] = (_Float16)x.y;
  dst[2] = (_Float16)x.z; dst[3] = (_Float16)x.w;
  dst[4] = (_Float16)y.x; dst[5] = (_Float16)y.y;
  dst[6] = (_Float16)y.z; dst[7] = (_Float16)y.w;
}

// A is a virtual [M, Ktot] fp32 matrix concatenated from up to 4 row-major
// sources (widths f0..f3, all multiples of 32 -> each 32-wide k-step stays in
// one region; region walk is UNIFORM scalar control flow, no EXEC divergence).
// BT is [128][Ktot] fp16 (transposed fused weights).
// Each block: 32 rows x 128 cols. Wave w owns col-tile 16*w for BOTH 16-row
// halves: two accumulators share one B fragment (2 v_wmma per k-step).
__global__ __launch_bounds__(256) void k_gemm_ln_mean(
    const float* __restrict__ s0, int f0, const float* __restrict__ s1, int f1,
    const float* __restrict__ s2, int f2, const float* __restrict__ s3, int f3,
    int Ktot, const _Float16* __restrict__ BT, const float* __restrict__ bias,
    float scale, const float* __restrict__ lng, const float* __restrict__ lnb,
    float* __restrict__ gsum) {
  __shared__ float tile[32][HID + 8];
  __shared__ float cs[HID];
  const int tid  = threadIdx.x;
  const int wave = tid >> 5, lane = tid & 31;
  const int hh   = lane >> 4, lr = lane & 15;
  const int m0   = blockIdx.x << 5;
  const int n0   = wave << 4;
  const int row0 = m0 + lr;
  const int row1 = m0 + 16 + lr;
  const _Float16* btcol = BT + (size_t)(n0 + lr) * Ktot + 16 * hh;

  v8f acc0 = {}, acc1 = {};
  int gk0 = 0;
  auto region = [&](const float* S, int F) {
    if (F == 0) return;
    const float* p0 = S + (size_t)row0 * F;
    const float* p1 = S + (size_t)row1 * F;
    for (int kk = 0; kk < F; kk += 32) {
      union { v16h v; _Float16 h[16]; } A0, A1;
      // 16-bit A 16x32 layout: halves j <- K=kk+16*hh+j (matches B)
      cvt8(A0.h,     p0 + kk + 16 * hh);
      cvt8(A0.h + 8, p0 + kk + 16 * hh + 8);
      cvt8(A1.h,     p1 + kk + 16 * hh);
      cvt8(A1.h + 8, p1 + kk + 16 * hh + 8);
      // B fragment: lane = column n0+lr, halves j <- B[gk0+kk+16*hh + j, col]
      v16h B = *(const v16h*)(btcol + gk0 + kk);
      acc0 = __builtin_amdgcn_wmma_f32_16x16x32_f16(false, A0.v, false, B,
                                                    (short)0, acc0, false, false);
      acc1 = __builtin_amdgcn_wmma_f32_16x16x32_f16(false, A1.v, false, B,
                                                    (short)0, acc1, false, false);
    }
    gk0 += F;
  };
  region(s0, f0);
  region(s1, f1);
  region(s2, f2);
  region(s3, f3);

  // Epilogue: D VGPR v holds (M = v + 8*hh, N = n0+lr)
  union { v8f v; float f[8]; } u0, u1;
  u0.v = acc0; u1.v = acc1;
  float bo = bias[n0 + lr];
#pragma unroll
  for (int v = 0; v < 8; ++v) {
    tile[v + 8 * hh][n0 + lr]      = (u0.f[v] + bo) * scale;
    tile[16 + v + 8 * hh][n0 + lr] = (u1.f[v] + bo) * scale;
  }
  if (tid < HID) cs[tid] = 0.0f;
  __syncthreads();

  // LayerNorm: each 16-lane half-wave owns one row (xor-shuffles stay inside
  // a half-wave); two passes cover the 32 rows.
#pragma unroll
  for (int rr = 0; rr < 2; ++rr) {
    int r = 2 * wave + hh + 16 * rr;
    float s = 0.f;
#pragma unroll
    for (int m = 0; m < 8; ++m) s += tile[r][lr + 16 * m];
    s += __shfl_xor(s, 1); s += __shfl_xor(s, 2);
    s += __shfl_xor(s, 4); s += __shfl_xor(s, 8);
    float mu = s * (1.0f / HID);
    float vs = 0.f;
#pragma unroll
    for (int m = 0; m < 8; ++m) { float d = tile[r][lr + 16 * m] - mu; vs += d * d; }
    vs += __shfl_xor(vs, 1); vs += __shfl_xor(vs, 2);
    vs += __shfl_xor(vs, 4); vs += __shfl_xor(vs, 8);
    float rstd = rsqrtf(vs * (1.0f / HID) + 1e-5f);
#pragma unroll
    for (int m = 0; m < 8; ++m) {
      int cc = lr + 16 * m;
      float y = (tile[r][cc] - mu) * rstd * lng[cc] + lnb[cc];
      y = fmaxf(y, 0.0f);
      atomicAdd(&cs[cc], y);
    }
  }
  __syncthreads();
  if (tid < HID) atomicAdd(&gsum[tid], cs[tid]);
}

// ---------------- head MLP ----------------

__global__ void k_head(const float* __restrict__ sfm, const float* __restrict__ stp,
                       const float* __restrict__ ssm, const float* __restrict__ gf,
                       const float* __restrict__ W1, const float* __restrict__ b1,
                       const float* __restrict__ W2, const float* __restrict__ b2,
                       float* __restrict__ out) {
  __shared__ float h1[64];
  int o = threadIdx.x;  // 64 threads
  float acc = b1[o];
  for (int i = 0; i < 448; ++i) {
    float hv;
    if (i < 128)      hv = sfm[i] * (1.0f / (float)N_FM);
    else if (i < 256) hv = stp[i - 128] * (1.0f / (float)N_TP);
    else if (i < 384) hv = ssm[i - 256] * (1.0f / (float)N_SM);
    else              hv = gf[i - 384];
    acc += hv * W1[i * 64 + o];
  }
  h1[o] = fmaxf(acc, 0.0f);
  __syncthreads();
  if (o == 0) {
    float s = b2[0];
    for (int i = 0; i < 64; ++i) s += h1[i] * W2[i];
    out[0] = s;
  }
}

// ---------------- host launcher ----------------

extern "C" void kernel_launch(void* const* d_in, const int* in_sizes, int n_in,
                              void* d_out, int out_size, void* d_ws, size_t ws_size,
                              hipStream_t stream) {
  const float* x_fm       = (const float*)d_in[0];
  const float* x_sm       = (const float*)d_in[1];
  const float* period_vol = (const float*)d_in[2];
  const float* gf         = (const float*)d_in[3];
  const float* pe_table   = (const float*)d_in[4];
  const int* qoq_src = (const int*)d_in[5];  const int* qoq_dst = (const int*)d_in[6];
  const int* bp_src  = (const int*)d_in[7];  const int* bp_dst  = (const int*)d_in[8];
  const int* cp_src  = (const int*)d_in[9];  const int* cp_dst  = (const int*)d_in[10];
  const int* cd_src  = (const int*)d_in[11]; const int* cd_dst  = (const int*)d_in[12];
  const int* rb_src  = (const int*)d_in[13]; const int* rb_dst  = (const int*)d_in[14];
  const float* qoq_Wl = (const float*)d_in[15]; const float* qoq_bl = (const float*)d_in[16];
  const float* qoq_Wr = (const float*)d_in[17];
  const float* bp_Wl  = (const float*)d_in[18]; const float* bp_bl  = (const float*)d_in[19];
  const float* bp_Wr  = (const float*)d_in[20];
  const float* cp_Wl  = (const float*)d_in[21]; const float* cp_bl  = (const float*)d_in[22];
  const float* cp_Wr  = (const float*)d_in[23];
  const float* cd_Wl  = (const float*)d_in[24]; const float* cd_bl  = (const float*)d_in[25];
  const float* cd_Wr  = (const float*)d_in[26];
  const float* rb_Wl  = (const float*)d_in[27]; const float* rb_bl  = (const float*)d_in[28];
  const float* rb_Wr  = (const float*)d_in[29];
  const float* ln_fm_g = (const float*)d_in[30]; const float* ln_fm_b = (const float*)d_in[31];
  const float* ln_tp_g = (const float*)d_in[32]; const float* ln_tp_b = (const float*)d_in[33];
  const float* ln_sm_g = (const float*)d_in[34]; const float* ln_sm_b = (const float*)d_in[35];
  const float* head_W1 = (const float*)d_in[36]; const float* head_b1 = (const float*)d_in[37];
  const float* head_W2 = (const float*)d_in[38]; const float* head_b2 = (const float*)d_in[39];

  char* base = (char*)d_ws;
  size_t off = 0;
  auto alloc = [&](size_t bytes) -> char* {
    char* p = base + off;
    off = (off + bytes + 255) & ~(size_t)255;
    return p;
  };
  float* pe       = (float*)alloc((size_t)N_TP * 32 * 4);
  char*  zbeg     = base + off;
  float* acc_qoq  = (float*)alloc((size_t)N_FM * 64 * 4);
  float* acc_cp   = (float*)alloc((size_t)N_FM * 32 * 4);
  float* acc_rb   = (float*)alloc((size_t)N_FM * 32 * 4);
  float* acc_bp   = (float*)alloc((size_t)N_TP * 64 * 4);
  float* acc_cd   = (float*)alloc((size_t)N_SM * 32 * 4);
  float* cnt_qoq  = (float*)alloc((size_t)N_FM * 4);
  float* cnt_cp   = (float*)alloc((size_t)N_FM * 4);
  float* cnt_rb   = (float*)alloc((size_t)N_FM * 4);
  float* cnt_bp   = (float*)alloc((size_t)N_TP * 4);
  float* cnt_cd   = (float*)alloc((size_t)N_SM * 4);
  float* sum_fm   = (float*)alloc(HID * 4);
  float* sum_tp   = (float*)alloc(HID * 4);
  float* sum_sm   = (float*)alloc(HID * 4);
  char*  zend     = base + off;
  _Float16* BfmT  = (_Float16*)alloc(128 * 192 * 2);
  float* bias_fm  = (float*)alloc(128 * 4);
  _Float16* BtpT  = (_Float16*)alloc(128 * 96 * 2);
  _Float16* BsmT  = (_Float16*)alloc(128 * 64 * 2);

  unsigned long long zn = (unsigned long long)(zend - zbeg) / 4ull;
  k_zero<<<8192, 256, 0, stream>>>((float*)zbeg, zn);
  k_pe<<<(N_TP * 32 + 255) / 256, 256, 0, stream>>>(pe_table, period_vol, pe);

  // edge scatter (segment sums + counts)
  k_edge<<<62500, 256, 0, stream>>>(x_fm, qoq_src, qoq_dst, 1000000, 4, acc_qoq, cnt_qoq);
  k_edge<<<62500, 256, 0, stream>>>(x_fm, bp_src,  bp_dst,  1000000, 4, acc_bp,  cnt_bp);
  k_edge<<<15625, 256, 0, stream>>>(pe,   cp_src,  cp_dst,   500000, 3, acc_cp,  cnt_cp);
  k_edge<<<15625, 256, 0, stream>>>(pe,   cd_src,  cd_dst,   500000, 3, acc_cd,  cnt_cd);
  k_edge<<<31250, 256, 0, stream>>>(pe,   rb_src,  rb_dst,  1000000, 3, acc_rb,  cnt_rb);

  // segment mean = sum / max(count,1)
  k_div<<<(N_FM * 64 + 255) / 256, 256, 0, stream>>>(acc_qoq, cnt_qoq, N_FM * 64, 6);
  k_div<<<(N_FM * 32 + 255) / 256, 256, 0, stream>>>(acc_cp,  cnt_cp,  N_FM * 32, 5);
  k_div<<<(N_FM * 32 + 255) / 256, 256, 0, stream>>>(acc_rb,  cnt_rb,  N_FM * 32, 5);
  k_div<<<(N_TP * 64 + 255) / 256, 256, 0, stream>>>(acc_bp,  cnt_bp,  N_TP * 64, 6);
  k_div<<<(N_SM * 32 + 255) / 256, 256, 0, stream>>>(acc_cd,  cnt_cd,  N_SM * 32, 5);

  // pack fused transposed fp16 weights
  k_pack_fm<<<96, 256, 0, stream>>>(qoq_Wl, cp_Wl, rb_Wl, qoq_Wr, cp_Wr, rb_Wr,
                                    qoq_bl, cp_bl, rb_bl, BfmT, bias_fm);
  k_pack2<<<48, 256, 0, stream>>>(bp_Wl, 64, bp_Wr, 32, BtpT);
  k_pack2<<<32, 256, 0, stream>>>(cd_Wl, 32, cd_Wr, 32, BsmT);

  // fused WMMA GEMM + bias + (1/3 for fm) + LayerNorm + ReLU + column-mean accum
  k_gemm_ln_mean<<<N_FM / 32, 256, 0, stream>>>(
      acc_qoq, 64, acc_cp, 32, acc_rb, 32, x_fm, 64,
      192, BfmT, bias_fm, 1.0f / 3.0f, ln_fm_g, ln_fm_b, sum_fm);
  k_gemm_ln_mean<<<N_TP / 32, 256, 0, stream>>>(
      acc_bp, 64, pe, 32, nullptr, 0, nullptr, 0,
      96, BtpT, bp_bl, 1.0f, ln_tp_g, ln_tp_b, sum_tp);
  k_gemm_ln_mean<<<N_SM / 32, 256, 0, stream>>>(
      acc_cd, 32, x_sm, 32, nullptr, 0, nullptr, 0,
      64, BsmT, cd_bl, 1.0f, ln_sm_g, ln_sm_b, sum_sm);

  k_head<<<1, 64, 0, stream>>>(sum_fm, sum_tp, sum_sm, gf,
                               head_W1, head_b1, head_W2, head_b2, (float*)d_out);
}
